// SimpleAttention_35227321762428
// MI455X (gfx1250) — matrix operand order repaired and checked by reading, not compiled, not run on hardware
//
#include <hip/hip_runtime.h>
#include <hip/hip_bf16.h>

// MI455X (gfx1250) fused single-query attention block.
// Key optimization: re-associate K/V projections around the single query row,
// turning 285 GFLOP of GEMM into ~16 GFLOP and making the launch HBM-bound on
// one pass over seq (536 MB -> ~23us at 23.3 TB/s). Dense math in bf16 WMMA;
// the seq stream is staged HBM->LDS by the Tensor Data Mover (no VGPR round-trip).

typedef __bf16 bf16_t;
typedef __attribute__((ext_vector_type(16))) __bf16 v16bf;
typedef __attribute__((ext_vector_type(8)))  float  v8f;
typedef unsigned int u32x4 __attribute__((ext_vector_type(4)));
typedef int i32x8 __attribute__((ext_vector_type(8)));
typedef int i32x4 __attribute__((ext_vector_type(4)));

#define BB 4096
#define NN 64
#define DD 512
#define DKK 256

__device__ __forceinline__ v8f wmma_bf16(v16bf a, v16bf b, v8f c) {
  return __builtin_amdgcn_wmma_f32_16x16x32_bf16(false, a, false, b, (short)0, c,
                                                 false, false);
}

// A-fragment (16x32 bf16, M x K). ISA layout: lanes 0-15 -> M rows, lane-half
// selects K sub-block: e<8 -> K = half*8+e ; e>=8 -> K = 8+half*8+e.
template <typename AT>
__device__ __forceinline__ v16bf load_a(const AT* A, int lda, int m0, int k0, int lane) {
  const int half = lane >> 4;
  const AT* row = A + (size_t)(m0 + (lane & 15)) * lda + k0;
  v16bf a;
#pragma unroll
  for (int e = 0; e < 8; ++e) a[e] = (bf16_t)(float)row[half * 8 + e];
#pragma unroll
  for (int e = 0; e < 8; ++e) a[8 + e] = (bf16_t)(float)row[16 + half * 8 + e];
  return a;
}

// B-fragment (32x16, K x N). lane&15 = N column, element = K (same order as A).
template <typename BT>
__device__ __forceinline__ v16bf load_b(const BT* Bw, int ldb, int k0, int n0, int lane) {
  const int half = lane >> 4;
  const BT* p = Bw + (size_t)k0 * ldb + n0 + (lane & 15);
  v16bf b;
#pragma unroll
  for (int e = 0; e < 8; ++e) b[e] = (bf16_t)(float)p[(size_t)(half * 8 + e) * ldb];
#pragma unroll
  for (int e = 0; e < 8; ++e) b[8 + e] = (bf16_t)(float)p[(size_t)(16 + half * 8 + e) * ldb];
  return b;
}

// C/D f32 16x16: element r at lane l -> M = r + 8*(l>=16), N = l&15.
// epi: 0 = f32 store, 1 = bf16 store, 2 = bias+relu -> bf16, 3 = bias -> f32
__device__ __forceinline__ void store_acc(v8f acc, void* C, const float* bias, int m0,
                                          int n0, int lane, int ldc, int epi) {
  const int n = n0 + (lane & 15);
  const int mb = m0 + 8 * (lane >> 4);
  const float bv = bias ? bias[n] : 0.0f;
#pragma unroll
  for (int r = 0; r < 8; ++r) {
    float v = acc[r];
    if (epi >= 2) v += bv;
    if (epi == 2 && v < 0.0f) v = 0.0f;
    const size_t idx = (size_t)(mb + r) * ldc + n;
    if (epi == 1 || epi == 2)
      ((bf16_t*)C)[idx] = (bf16_t)v;
    else
      ((float*)C)[idx] = v;
  }
}

// ---------------------------------------------------------------------------
// Generic GEMM: C[M,N] = A[M,K] @ Bw[K,N]   (block tile 64x64, wave tile 32x16)
// grid = (N/64, M/64), block = 256 threads (8 waves).
// ---------------------------------------------------------------------------
template <typename AT>
__global__ void gemm_wmma_kernel(const AT* __restrict__ A, const bf16_t* __restrict__ Bw,
                                 const float* __restrict__ bias, void* __restrict__ C,
                                 int K, int lda, int ldb, int ldc, int epi) {
  const int lane = threadIdx.x & 31;
  const int wave = threadIdx.x >> 5;
  const int wm = wave >> 2, wn = wave & 3;
  const int m0 = blockIdx.y * 64 + wm * 32;
  const int n0 = blockIdx.x * 64 + wn * 16;
  v8f acc0 = {}, acc1 = {};
  for (int k0 = 0; k0 < K; k0 += 32) {
    v16bf a0 = load_a<AT>(A, lda, m0, k0, lane);
    v16bf a1 = load_a<AT>(A, lda, m0 + 16, k0, lane);
    v16bf bb = load_b<bf16_t>(Bw, ldb, k0, n0, lane);
    acc0 = wmma_bf16(a0, bb, acc0);
    acc1 = wmma_bf16(a1, bb, acc1);
  }
  store_acc(acc0, C, bias, m0, n0, lane, ldc, epi);
  store_acc(acc1, C, bias, m0 + 16, n0, lane, ldc, epi);
}

// ---------------------------------------------------------------------------
// Weight prep: bf16 copies of Wq, W1, W2 and per-head transposed Wk:
//   wkt[h][d*512 + c] = Wk[c*512 + h*256 + d]
// ---------------------------------------------------------------------------
__global__ void prep_weights_kernel(const float* __restrict__ Wq, const float* __restrict__ Wk,
                                    const float* __restrict__ W1, const float* __restrict__ W2,
                                    bf16_t* __restrict__ wq_bf, bf16_t* __restrict__ wkt_bf,
                                    bf16_t* __restrict__ w1_bf, bf16_t* __restrict__ w2_bf) {
  const int i = blockIdx.x * blockDim.x + threadIdx.x;  // 0 .. 262143
  if (i >= 262144) return;
  wq_bf[i] = (bf16_t)Wq[i];
  w2_bf[i] = (bf16_t)W2[i];
  w1_bf[i] = (bf16_t)W1[i];
  w1_bf[i + 262144] = (bf16_t)W1[i + 262144];
  const int h = i >> 17;
  const int d = (i >> 9) & 255;
  const int c = i & 511;
  wkt_bf[i] = (bf16_t)Wk[c * 512 + h * 256 + d];
}

// ---------------------------------------------------------------------------
// Wvo[h*512 + c, e] = sum_d Wv[c, h*256+d] * Wo[h*256+d, e]   (WMMA, f32 in)
// M = 1024, N = 512, K = 256.  grid = (8, 16).
// ---------------------------------------------------------------------------
__device__ __forceinline__ v16bf load_a_wv(const float* Wv, int m0, int k0, int lane, int h) {
  const int m = m0 + (lane & 15);
  const int c = m & 511;
  const int half = lane >> 4;
  const float* row = Wv + (size_t)c * 512 + h * 256 + k0;
  v16bf a;
#pragma unroll
  for (int e = 0; e < 8; ++e) a[e] = (bf16_t)row[half * 8 + e];
#pragma unroll
  for (int e = 0; e < 8; ++e) a[8 + e] = (bf16_t)row[16 + half * 8 + e];
  return a;
}

__global__ void prep_wvo_kernel(const float* __restrict__ Wv, const float* __restrict__ Wo,
                                bf16_t* __restrict__ wvo) {
  const int lane = threadIdx.x & 31;
  const int wave = threadIdx.x >> 5;
  const int wm = wave >> 2, wn = wave & 3;
  const int m0 = blockIdx.y * 64 + wm * 32;
  const int n0 = blockIdx.x * 64 + wn * 16;
  const int h = m0 >> 9;  // 64-row block never crosses the 512-row head boundary
  v8f acc0 = {}, acc1 = {};
  for (int k0 = 0; k0 < 256; k0 += 32) {
    v16bf a0 = load_a_wv(Wv, m0, k0, lane, h);
    v16bf a1 = load_a_wv(Wv, m0 + 16, k0, lane, h);
    v16bf bb = load_b<float>(Wo + (size_t)h * 256 * 512, 512, k0, n0, lane);
    acc0 = wmma_bf16(a0, bb, acc0);
    acc1 = wmma_bf16(a1, bb, acc1);
  }
  store_acc(acc0, wvo, nullptr, m0, n0, lane, 512, 1);
  store_acc(acc1, wvo, nullptr, m0 + 16, n0, lane, 512, 1);
}

// ---------------------------------------------------------------------------
// kq[b, h*512 + c] = sum_d qh[b, h*256+d] * wkt[h][d*512 + c]
// M = 4096, N = 1024, K = 256.  grid = (16, 64).
// ---------------------------------------------------------------------------
__global__ void gemm_kq_kernel(const bf16_t* __restrict__ qh, const bf16_t* __restrict__ wkt,
                               float* __restrict__ kq) {
  const int lane = threadIdx.x & 31;
  const int wave = threadIdx.x >> 5;
  const int wm = wave >> 2, wn = wave & 3;
  const int m0 = blockIdx.y * 64 + wm * 32;
  const int n0 = blockIdx.x * 64 + wn * 16;   // global column 0..1023
  const int h = n0 >> 9;                       // head from column
  const bf16_t* A = qh + h * 256;              // lda = 512
  const bf16_t* Bm = wkt + (size_t)h * 256 * 512 + (n0 & 511);
  v8f acc0 = {}, acc1 = {};
  for (int k0 = 0; k0 < 256; k0 += 32) {
    v16bf a0 = load_a<bf16_t>(A, 512, m0, k0, lane);
    v16bf a1 = load_a<bf16_t>(A, 512, m0 + 16, k0, lane);
    v16bf bb = load_b<bf16_t>(Bm, 512, k0, 0, lane);
    acc0 = wmma_bf16(a0, bb, acc0);
    acc1 = wmma_bf16(a1, bb, acc1);
  }
  store_acc(acc0, kq, nullptr, m0, n0, lane, 1024, 0);
  store_acc(acc1, kq, nullptr, m0 + 16, n0, lane, 1024, 0);
}

// ---------------------------------------------------------------------------
// Fused attention: one workgroup per batch element.  seq[b] (64x512 f32,
// 128 KB) is DMA'd HBM->LDS by the Tensor Data Mover (one descriptor per WG,
// issued by wave 0, tracked by TENSORcnt) and reused for scores AND ctx ->
// single HBM pass over seq, zero VGPR staging traffic.
//   scores[h,n] = (1/16) * seq[n,:] . kq[b,h,:]   (+ mask -> -1e10)
//   attn = softmax ; weight[b,n] = mean_h attn ; ctx[h,c] = sum_n attn*seq[n,c]
// ---------------------------------------------------------------------------
#define ATTN_SMEM (64 * 512 * 4 + 1024 * 4 + 256 * 4 + 128 * 4 + 128 * 4 + 16)

__global__ void attn_kernel(const float* __restrict__ seq, const float* __restrict__ kq,
                            const unsigned char* __restrict__ mask,
                            float* __restrict__ weight_out, bf16_t* __restrict__ ctx_out) {
  extern __shared__ char smem[];
  float* sseq = (float*)smem;                          // 64*512 f32 (TDM target)
  float* skq = (float*)(smem + 64 * 512 * 4);          // 1024
  float* sred = skq + 1024;                            // 256
  float* sscore = sred + 256;                          // 128
  float* sattn = sscore + 128;                         // 128
  float* ssm = sattn + 128;                            // 4
  const int b = blockIdx.x;
  const int t = threadIdx.x;

  // --- TDM: DMA seq[b] (2-D tile 512 x 64 of 4-byte elems) into LDS ---------
  if (t < 32) {
    const unsigned long long ga = (unsigned long long)(seq + (size_t)b * 64 * 512);
    const unsigned lds_addr = (unsigned)(size_t)sseq;  // flat LDS addr, low 32 bits
    // D# group0: count=1, lds_addr, global_addr[56:0], type=2 (bits 127:126)
    u32x4 g0 = {1u, lds_addr, (unsigned)ga, (unsigned)(ga >> 32) | 0x80000000u};
    // D# group1: data_size=4B (bits 17:16 = 2); tensor_dim0 = 512 (bits 79:48),
    // tensor_dim1 = 64 (bits 111:80), tile_dim0 = 512 (127:112),
    // tile_dim1 = 64 (143:128), tensor_dim0_stride = 512 (207:160).
    i32x8 g1 = {0x20000, 512 << 16, 64 << 16, 512 << 16, 64, 512, 0, 0};
    i32x4 gz = {0, 0, 0, 0};
#if defined(__clang_major__) && __clang_major__ >= 23
    i32x8 gz8 = {0, 0, 0, 0, 0, 0, 0, 0};
    __builtin_amdgcn_tensor_load_to_lds(g0, g1, gz, gz, gz8, 0);
#else
    __builtin_amdgcn_tensor_load_to_lds(g0, g1, gz, gz, 0);
#endif
  }
  // overlap: all threads pull kq row while the TDM streams
  for (int i = t; i < 1024; i += 256) skq[i] = kq[(size_t)b * 1024 + i];
  if (t < 32) __builtin_amdgcn_s_wait_tensorcnt(0);
  __syncthreads();

  const int n = t & 63;
  const int part = t >> 6;  // 4 partial-sum groups per score
  for (int h = 0; h < 2; ++h) {
    const float* kqh = skq + h * 512;
    float p = 0.0f;
    for (int j = 0; j < 128; ++j) {
      const int c = part * 128 + ((j + n) & 127);  // stagger: avoid bank conflicts
      p += sseq[n * 512 + c] * kqh[c];
    }
    sred[t] = p;
    __syncthreads();
    if (t < 64) {
      float s = sred[n] + sred[64 + n] + sred[128 + n] + sred[192 + n];
      s *= 0.0625f;  // 1/sqrt(DK)=1/16
      if (mask[(size_t)b * 64 + n]) s = -1.0e10f;
      sscore[h * 64 + n] = s;
    }
    __syncthreads();
  }

  if (t < 2) {  // per-head max & 1/sum (64-wide, trivially cheap)
    float mx = -3.4e38f;
    for (int i = 0; i < 64; ++i) mx = fmaxf(mx, sscore[t * 64 + i]);
    float sm = 0.0f;
    for (int i = 0; i < 64; ++i) sm += __expf(sscore[t * 64 + i] - mx);
    ssm[t * 2 + 0] = mx;
    ssm[t * 2 + 1] = 1.0f / sm;
  }
  __syncthreads();
  if (t < 128) {
    const int h = t >> 6;
    sattn[t] = __expf(sscore[t] - ssm[h * 2]) * ssm[h * 2 + 1];
  }
  __syncthreads();
  if (t < 64) weight_out[(size_t)b * 64 + t] = 0.5f * (sattn[t] + sattn[64 + t]);

  // ctx[h,c] = sum_n attn[h,n] * seq[n,c]   (consecutive c -> conflict-free)
  for (int h = 0; h < 2; ++h) {
#pragma unroll
    for (int cb = 0; cb < 512; cb += 256) {
      const int c = cb + t;
      float acc = 0.0f;
      for (int m = 0; m < 64; ++m) acc += sattn[h * 64 + m] * sseq[m * 512 + c];
      ctx_out[(size_t)b * 1024 + h * 512 + c] = (bf16_t)acc;
    }
  }
}

// ---------------------------------------------------------------------------
// out0 = ctx @ Wvo + bo + src ; LayerNorm ; xout = [ln_out | src] in bf16.
// One block per 16 rows, full 512-wide row in-block for the LN reduction.
// ---------------------------------------------------------------------------
__global__ void out_ln_kernel(const bf16_t* __restrict__ ctx, const bf16_t* __restrict__ wvo,
                              const float* __restrict__ bo, const float* __restrict__ src,
                              const float* __restrict__ g, const float* __restrict__ bt,
                              bf16_t* __restrict__ xout) {
  __shared__ float tile[16][520];
  __shared__ float p1[16][16], p2[16][16];
  __shared__ float stat[16][2];
  const int lane = threadIdx.x & 31;
  const int wave = threadIdx.x >> 5;
  const int m0 = blockIdx.x * 16;
  v8f acc[4] = {};
  for (int k0 = 0; k0 < 1024; k0 += 32) {
    v16bf a = load_a<bf16_t>(ctx, 1024, m0, k0, lane);
#pragma unroll
    for (int s = 0; s < 4; ++s) {
      v16bf bb = load_b<bf16_t>(wvo, 512, k0, wave * 64 + s * 16, lane);
      acc[s] = wmma_bf16(a, bb, acc[s]);
    }
  }
#pragma unroll
  for (int s = 0; s < 4; ++s) {
    const int nc = wave * 64 + s * 16 + (lane & 15);
    const int mb = 8 * (lane >> 4);
#pragma unroll
    for (int r = 0; r < 8; ++r) tile[mb + r][nc] = acc[s][r];
  }
  __syncthreads();

  const int r = threadIdx.x >> 4;
  const int idx = threadIdx.x & 15;
  const int row = m0 + r;
  float s1 = 0.0f, s2 = 0.0f;
  for (int e = idx * 32; e < idx * 32 + 32; ++e) {
    const float v = tile[r][e] + bo[e] + src[(size_t)row * 512 + e];
    tile[r][e] = v;
    s1 += v;
    s2 += v * v;
  }
  p1[r][idx] = s1;
  p2[r][idx] = s2;
  __syncthreads();
  if (idx == 0) {
    float a1 = 0.0f, a2 = 0.0f;
    for (int i = 0; i < 16; ++i) { a1 += p1[r][i]; a2 += p2[r][i]; }
    const float mu = a1 * (1.0f / 512.0f);
    const float var = a2 * (1.0f / 512.0f) - mu * mu;
    stat[r][0] = mu;
    stat[r][1] = rsqrtf(var + 1e-5f);
  }
  __syncthreads();
  const float mu = stat[r][0], rs = stat[r][1];
  for (int e = idx * 32; e < idx * 32 + 32; ++e) {
    const float v = (tile[r][e] - mu) * rs * g[e] + bt[e];
    xout[(size_t)row * 1024 + e] = (bf16_t)v;
    xout[(size_t)row * 1024 + 512 + e] = (bf16_t)src[(size_t)row * 512 + e];
  }
}

// ---------------------------------------------------------------------------
extern "C" void kernel_launch(void* const* d_in, const int* in_sizes, int n_in,
                              void* d_out, int out_size, void* d_ws, size_t ws_size,
                              hipStream_t stream) {
  const float* src = (const float*)d_in[0];
  const float* seq = (const float*)d_in[1];
  const unsigned char* mask = (const unsigned char*)d_in[2];  // jax bool_ = 1 byte
  const float* Wq = (const float*)d_in[3];
  const float* Wk = (const float*)d_in[4];
  const float* Wv = (const float*)d_in[5];
  const float* Wo = (const float*)d_in[6];
  const float* bo = (const float*)d_in[7];
  const float* ln_g = (const float*)d_in[8];
  const float* ln_b = (const float*)d_in[9];
  const float* W1 = (const float*)d_in[10];
  const float* b1 = (const float*)d_in[11];
  const float* W2 = (const float*)d_in[12];
  const float* b2 = (const float*)d_in[13];

  float* y = (float*)d_out;                        // (B, 512)
  float* weight = y + (size_t)BB * DD;             // (B, 64)

  char* p = (char*)d_ws;
  bf16_t* wq_bf = (bf16_t*)p;  p += (size_t)512 * 512 * 2;
  bf16_t* wkt_bf = (bf16_t*)p; p += (size_t)2 * 256 * 512 * 2;
  bf16_t* wvo_bf = (bf16_t*)p; p += (size_t)1024 * 512 * 2;
  bf16_t* w1_bf = (bf16_t*)p;  p += (size_t)1024 * 512 * 2;
  bf16_t* w2_bf = (bf16_t*)p;  p += (size_t)512 * 512 * 2;
  bf16_t* qh_bf = (bf16_t*)p;  p += (size_t)BB * 512 * 2;
  float* kq = (float*)p;       p += (size_t)BB * 1024 * 4;
  bf16_t* ctx_bf = (bf16_t*)p; p += (size_t)BB * 1024 * 2;
  bf16_t* x_bf = (bf16_t*)p;   p += (size_t)BB * 1024 * 2;
  bf16_t* h_bf = (bf16_t*)p;   p += (size_t)BB * 512 * 2;

  // 1. bf16 weight prep (+ per-head Wk transpose)
  prep_weights_kernel<<<1024, 256, 0, stream>>>(Wq, Wk, W1, W2, wq_bf, wkt_bf, w1_bf, w2_bf);
  // 2. Wvo_h = Wv_h @ Wo_h  (folds the output projection)
  prep_wvo_kernel<<<dim3(8, 16), 256, 0, stream>>>(Wv, Wo, wvo_bf);
  // 3. qh = src @ Wq  -> bf16
  gemm_wmma_kernel<float><<<dim3(8, 64), 256, 0, stream>>>(src, wq_bf, nullptr, qh_bf,
                                                           512, 512, 512, 512, 1);
  // 4. kq[b,h,:] = Wk_h^T-contraction of qh  -> f32
  gemm_kq_kernel<<<dim3(16, 64), 256, 0, stream>>>(qh_bf, wkt_bf, kq);
  // 5. fused attention: TDM-staged single HBM pass over seq (the 536 MB term)
  attn_kernel<<<BB, 256, ATTN_SMEM, stream>>>(seq, kq, mask, weight, ctx_bf);
  // 6. ctx @ Wvo + bo + residual, LayerNorm, concat with src
  out_ln_kernel<<<256, 256, 0, stream>>>(ctx_bf, wvo_bf, bo, src, ln_g, ln_b, x_bf);
  // 7. FFN1: relu(x @ W1 + b1) -> bf16
  gemm_wmma_kernel<bf16_t><<<dim3(8, 64), 256, 0, stream>>>(x_bf, w1_bf, b1, h_bf,
                                                            1024, 1024, 512, 512, 2);
  // 8. FFN2: y = h @ W2 + b2 -> f32 output
  gemm_wmma_kernel<bf16_t><<<dim3(8, 64), 256, 0, stream>>>(h_bf, w2_bf, b2, y,
                                                            512, 512, 512, 512, 3);
  (void)in_sizes; (void)n_in; (void)out_size; (void)ws_size;
}
